// KLayerMambaModel_33861522162490
// MI455X (gfx1250) — matrix operand order repaired and checked by reading, not compiled, not run on hardware
//
#include <hip/hip_runtime.h>
#include <math.h>

#define NT   1024
#define NL   2
#define DM   768
#define DS   16
#define DC   4
#define DI   1536
#define DR   48
#define DRP  64          // dt rank padded so K%32==0
#define GDIM 80          // DR + 2*DS
#define B_   2
#define L_   2048
#define MTOT (B_ * L_)   // 4096

typedef __attribute__((ext_vector_type(16))) __bf16 v16bf;
typedef __attribute__((ext_vector_type(8)))  __bf16 v8bf;
typedef __attribute__((ext_vector_type(8)))  float  v8f;

union abfrag { v16bf v; v8bf h[2]; };

// ---------------------------------------------------------------------------
// C[M,N] = A[M,K] @ W[N,K]^T (+bias). A,W are bf16 in memory, fp32 accumulate.
// One wave computes a 32 x (16*NW) tile: 2 A-frags x NW B-frags -> 2*NW WMMAs
// per 32-wide K-step (12 b128 loads / 8 WMMAs at NW=4). 8 waves / block.
// Fragment layouts per CDNA5 ISA 7.12.2 (wave32):
//   A 16x32 bf16 : lane m=L&15, half=L>>4; elems 0..7 -> K=half*8+e,
//                  elems 8..15 -> K=16+half*8+(e-8)
//   B 32x16 bf16 : same K mapping as A (lane n=L&15)
//   C/D 16x16 f32: lane n=L&15; vgpr r -> row m=half*8+r
// Requires M%32==0, N%(16*NW)==0, K%32==0 (true at all call sites).
// ---------------------------------------------------------------------------
template <int NW>
__global__ __launch_bounds__(256) void gemm_bf16(
    const __bf16* __restrict__ A, const __bf16* __restrict__ W,
    float* __restrict__ Cf, __bf16* __restrict__ Cb,
    const float* __restrict__ bias, int M, int N, int K)
{
  const int wave = threadIdx.x >> 5;
  const int lane = threadIdx.x & 31;
  const int hf   = lane >> 4;
  const int l16  = lane & 15;
  const int tm   = blockIdx.y * 8 + wave;           // 32-row tile index
  if (tm * 32 >= M) return;                         // wave-uniform exit
  const int n0 = blockIdx.x * (16 * NW) + l16;

  const __bf16* Ar0 = A + (size_t)(tm * 32 + l16) * K;
  const __bf16* Ar1 = Ar0 + (size_t)16 * K;
  const __bf16* Wp[NW];
#pragma unroll
  for (int j = 0; j < NW; ++j) Wp[j] = W + (size_t)(n0 + 16 * j) * K;

  const v8f z8 = {0.f, 0.f, 0.f, 0.f, 0.f, 0.f, 0.f, 0.f};
  v8f acc[2][NW];
#pragma unroll
  for (int i = 0; i < 2; ++i)
#pragma unroll
    for (int j = 0; j < NW; ++j) acc[i][j] = z8;

  for (int k0 = 0; k0 < K; k0 += 32) {
    if (k0 + 32 < K) {
      __builtin_prefetch(Ar0 + k0 + 32, 0, 3);      // global_prefetch_b8
      __builtin_prefetch(Ar1 + k0 + 32, 0, 3);
#pragma unroll
      for (int j = 0; j < NW; ++j) __builtin_prefetch(Wp[j] + k0 + 32, 0, 3);
    }
    abfrag a0, a1;
    a0.h[0] = *(const v8bf*)(Ar0 + k0 + hf * 8);
    a0.h[1] = *(const v8bf*)(Ar0 + k0 + 16 + hf * 8);
    a1.h[0] = *(const v8bf*)(Ar1 + k0 + hf * 8);
    a1.h[1] = *(const v8bf*)(Ar1 + k0 + 16 + hf * 8);
    v16bf b[NW];
#pragma unroll
    for (int j = 0; j < NW; ++j) { abfrag t; t.h[0] = *(const v8bf*)(Wp[j] + k0 + hf * 8); t.h[1] = *(const v8bf*)(Wp[j] + k0 + 16 + hf * 8); b[j] = t.v; }
#pragma unroll
    for (int j = 0; j < NW; ++j) {
      acc[0][j] = __builtin_amdgcn_wmma_f32_16x16x32_bf16(
          false, a0.v, false, b[j], (short)0, acc[0][j], false, false);
      acc[1][j] = __builtin_amdgcn_wmma_f32_16x16x32_bf16(
          false, a1.v, false, b[j], (short)0, acc[1][j], false, false);
    }
  }

#pragma unroll
  for (int j = 0; j < NW; ++j) {
    const int nc = n0 + 16 * j;
    const float bv = bias ? bias[nc] : 0.f;
#pragma unroll
    for (int i = 0; i < 2; ++i) {
#pragma unroll
      for (int r = 0; r < 8; ++r) {
        const int m = tm * 32 + i * 16 + hf * 8 + r;
        const float v = acc[i][j][r] + bv;
        if (Cf) Cf[(size_t)m * N + nc] = v;
        if (Cb) Cb[(size_t)m * N + nc] = (__bf16)v;
      }
    }
  }
}

// ---------------------------------------------------------------------------
__global__ void f32_to_bf16_kernel(const float* __restrict__ src,
                                   __bf16* __restrict__ dst, int n)
{
  const int gid = blockIdx.x * blockDim.x + threadIdx.x;
  if (gid < n) dst[gid] = (__bf16)src[gid];
}

// h_bf[m,d] = bf16( Wi[d, x[m]] + bi[d] )
__global__ void embed_kernel(const int* __restrict__ x, const float* __restrict__ Wi,
                             const float* __restrict__ bi, __bf16* __restrict__ hbf)
{
  const int m   = blockIdx.x;
  const int tok = x[m];
  for (int d = threadIdx.x; d < DM; d += blockDim.x)
    hbf[(size_t)m * DM + d] = (__bf16)(Wi[(size_t)d * NT + tok] + bi[d]);
}

// u = silu(causal_conv(xz[:, :DI])); writes fp32 (for scan) + bf16 (for GEMM)
__global__ void conv_silu_kernel(const float* __restrict__ xz, const float* __restrict__ cw,
                                 const float* __restrict__ cb,
                                 float* __restrict__ usil, __bf16* __restrict__ usilbf)
{
  const int gid = blockIdx.x * blockDim.x + threadIdx.x;
  if (gid >= MTOT * DI) return;
  const int m = gid / DI, d = gid - m * DI;
  const int t = m & (L_ - 1);
  float acc = cb[d];
#pragma unroll
  for (int k = 0; k < DC; ++k) {
    const int tt = t - (DC - 1) + k;
    if (tt >= 0)
      acc += xz[(size_t)(m - (DC - 1) + k) * (2 * DI) + d] * cw[d * DC + k];
  }
  const float s = acc / (1.f + expf(-acc));
  usil[gid]   = s;
  usilbf[gid] = (__bf16)s;
}

// dtpad_bf[m, 0..63] = { bf16(xdbl[m, 0..47]), 0... }
__global__ void pad_dt_act(const float* __restrict__ xdbl, __bf16* __restrict__ dtpad)
{
  const int gid = blockIdx.x * blockDim.x + threadIdx.x;
  if (gid >= MTOT * DRP) return;
  const int m = gid / DRP, c = gid - m * DRP;
  dtpad[gid] = (c < DR) ? (__bf16)xdbl[(size_t)m * GDIM + c] : (__bf16)0.f;
}

// dtw_bf[n, 0..63] = { bf16(dt_w[n, 0..47]), 0... }
__global__ void pad_dt_w(const float* __restrict__ w, __bf16* __restrict__ wp)
{
  const int gid = blockIdx.x * blockDim.x + threadIdx.x;
  if (gid >= DI * DRP) return;
  const int n = gid / DRP, c = gid - n * DRP;
  wp[gid] = (c < DR) ? (__bf16)w[(size_t)n * DR + c] : (__bf16)0.f;
}

// buf = softplus(buf + dt_b[d])   (numerically stable)
__global__ void softplus_bias_kernel(float* __restrict__ buf, const float* __restrict__ dtb)
{
  const int gid = blockIdx.x * blockDim.x + threadIdx.x;
  if (gid >= MTOT * DI) return;
  const int d = gid % DI;
  const float v = buf[gid] + dtb[d];
  buf[gid] = fmaxf(v, 0.f) + log1pf(expf(-fabsf(v)));
}

// ---------------------------------------------------------------------------
// Selective scan, one thread per (b,d) channel, DS=16 states in registers.
// y = (scan + u*Dp) * silu(z) -> written as bf16 for the out-proj GEMM.
// ---------------------------------------------------------------------------
__global__ void scan_kernel(const float* __restrict__ delta, const float* __restrict__ usil,
                            const float* __restrict__ xz, const float* __restrict__ xdbl,
                            const float* __restrict__ Alog, const float* __restrict__ Dpl,
                            __bf16* __restrict__ ybf)
{
  const int gid = blockIdx.x * blockDim.x + threadIdx.x;
  if (gid >= B_ * DI) return;
  const int b = gid / DI, d = gid - b * DI;
  float Ac[DS], hs[DS];
#pragma unroll
  for (int n = 0; n < DS; ++n) { Ac[n] = -expf(Alog[d * DS + n]); hs[n] = 0.f; }
  const float dp = Dpl[d];
  for (int t = 0; t < L_; ++t) {
    const size_t m  = (size_t)b * L_ + t;
    const float  dl = delta[m * DI + d];
    const float  uu = usil[m * DI + d];
    const float  du = dl * uu;
    const float* bc = xdbl + m * GDIM + DR;     // Bm at [0..15], Cm at [16..31]
    float y = 0.f;
#pragma unroll
    for (int n = 0; n < DS; ++n) {
      hs[n] = expf(dl * Ac[n]) * hs[n] + du * bc[n];
      y += hs[n] * bc[DS + n];
    }
    const float zz = xz[m * (2 * DI) + DI + d];
    y = (y + uu * dp) * (zz / (1.f + expf(-zz)));
    ybf[m * DI + d] = (__bf16)y;
  }
}

// ---------------------------------------------------------------------------
static inline void launch_gemm(const __bf16* A, const __bf16* W, float* Cf, __bf16* Cb,
                               const float* bias, int M, int N, int K, hipStream_t s)
{
  const dim3 blk(256);
  const int gy = (M / 32 + 7) / 8;
  if (N % 64 == 0) {
    gemm_bf16<4><<<dim3(N / 64, gy), blk, 0, s>>>(A, W, Cf, Cb, bias, M, N, K);
  } else if (N % 32 == 0) {
    gemm_bf16<2><<<dim3(N / 32, gy), blk, 0, s>>>(A, W, Cf, Cb, bias, M, N, K);
  } else {
    gemm_bf16<1><<<dim3(N / 16, gy), blk, 0, s>>>(A, W, Cf, Cb, bias, M, N, K);
  }
}

static inline void cvt(const float* s, __bf16* d, int n, hipStream_t st)
{
  f32_to_bf16_kernel<<<(n + 255) / 256, 256, 0, st>>>(s, d, n);
}

extern "C" void kernel_launch(void* const* d_in, const int* in_sizes, int n_in,
                              void* d_out, int out_size, void* d_ws, size_t ws_size,
                              hipStream_t stream)
{
  (void)in_sizes; (void)n_in; (void)out_size; (void)ws_size;
  const int*   x      = (const int*)  d_in[0];
  const float* Wi     = (const float*)d_in[1];
  const float* bi     = (const float*)d_in[2];
  const float* Wo     = (const float*)d_in[3];
  const float* bo     = (const float*)d_in[4];
  const float* in_w   = (const float*)d_in[5];
  const float* conv_w = (const float*)d_in[6];
  const float* conv_b = (const float*)d_in[7];
  const float* xprojw = (const float*)d_in[8];
  const float* dt_w   = (const float*)d_in[9];
  const float* dt_b   = (const float*)d_in[10];
  const float* A_log  = (const float*)d_in[11];
  const float* Dp     = (const float*)d_in[12];
  const float* out_w  = (const float*)d_in[13];
  float* out = (float*)d_out;

  // ---- workspace partition: fp32 region then bf16 region (~150 MB total) ----
  float* xz    = (float*)d_ws;                        // M x 2*DI
  float* usil  = xz    + (size_t)MTOT * 2 * DI;       // M x DI
  float* xdbl  = usil  + (size_t)MTOT * DI;           // M x 80
  float* delta = xdbl  + (size_t)MTOT * GDIM;         // M x DI
  __bf16* bfbase = (__bf16*)(delta + (size_t)MTOT * DI);
  __bf16* hbf    = bfbase;                            // M x DM
  __bf16* usilbf = hbf    + (size_t)MTOT * DM;        // M x DI
  __bf16* dtpad  = usilbf + (size_t)MTOT * DI;        // M x 64
  __bf16* ybf    = dtpad  + (size_t)MTOT * DRP;       // M x DI
  __bf16* inwbf  = ybf    + (size_t)MTOT * DI;        // NL x 2DI x DM
  __bf16* xpwbf  = inwbf  + (size_t)NL * 2 * DI * DM; // NL x 80 x DI
  __bf16* outwbf = xpwbf  + (size_t)NL * GDIM * DI;   // NL x DM x DI
  __bf16* Wobf   = outwbf + (size_t)NL * DM * DI;     // NT x DM
  __bf16* dtwbf  = Wobf   + (size_t)NT * DM;          // DI x 64 (per-layer reuse)

  // ---- one-time (per launch) weight conversions to bf16 ----
  cvt(in_w,   inwbf,  NL * 2 * DI * DM, stream);
  cvt(xprojw, xpwbf,  NL * GDIM * DI,   stream);
  cvt(out_w,  outwbf, NL * DM * DI,     stream);
  cvt(Wo,     Wobf,   NT * DM,          stream);

  embed_kernel<<<MTOT, 256, 0, stream>>>(x, Wi, bi, hbf);

  for (int l = 0; l < NL; ++l) {
    const __bf16* inw_l  = inwbf  + (size_t)l * 2 * DI * DM;
    const __bf16* xpw_l  = xpwbf  + (size_t)l * GDIM * DI;
    const __bf16* outw_l = outwbf + (size_t)l * DM * DI;
    const float*  cw_l   = conv_w + (size_t)l * DI * DC;
    const float*  cb_l   = conv_b + (size_t)l * DI;
    const float*  dtw_l  = dt_w   + (size_t)l * DI * DR;
    const float*  dtb_l  = dt_b   + (size_t)l * DI;
    const float*  Alog_l = A_log  + (size_t)l * DI * DS;
    const float*  Dp_l   = Dp     + (size_t)l * DI;

    // xz = h @ in_w^T                        (4096 x 3072 x 768), NW=4
    launch_gemm(hbf, inw_l, xz, nullptr, nullptr, MTOT, 2 * DI, DM, stream);
    // u = silu(causal_conv(xz[:, :DI]))
    conv_silu_kernel<<<(MTOT * DI + 255) / 256, 256, 0, stream>>>(xz, cw_l, cb_l, usil, usilbf);
    // xdbl = u @ xproj_w^T                   (4096 x 80 x 1536), NW=1 path
    launch_gemm(usilbf, xpw_l, xdbl, nullptr, nullptr, MTOT, GDIM, DI, stream);
    // dt-proj with K padded 48 -> 64
    pad_dt_act<<<(MTOT * DRP + 255) / 256, 256, 0, stream>>>(xdbl, dtpad);
    pad_dt_w<<<(DI * DRP + 255) / 256, 256, 0, stream>>>(dtw_l, dtwbf);
    launch_gemm(dtpad, dtwbf, delta, nullptr, nullptr, MTOT, DI, DRP, stream);
    // delta = softplus(delta + dt_b)
    softplus_bias_kernel<<<(MTOT * DI + 255) / 256, 256, 0, stream>>>(delta, dtb_l);
    // selective scan -> y (bf16)
    scan_kernel<<<(B_ * DI + 255) / 256, 256, 0, stream>>>(delta, usil, xz, xdbl, Alog_l, Dp_l, ybf);
    // h = y @ out_w^T  (bf16 epilogue only)  (4096 x 768 x 1536), NW=4
    launch_gemm(ybf, outw_l, nullptr, hbf, nullptr, MTOT, DM, DI, stream);
  }

  // logits = h @ Wo^T + bo                   (4096 x 1024 x 768), NW=4
  launch_gemm(hbf, Wobf, out, nullptr, bo, MTOT, NT, DM, stream);
}
